// kernel_residual_fc_embeds_48180943126694
// MI455X (gfx1250) — matrix operand order repaired and checked by reading, not compiled, not run on hardware
//
#include <hip/hip_runtime.h>
#include <hip/hip_bf16.h>
#include <math.h>

typedef __attribute__((ext_vector_type(16))) __bf16 v16bf;
typedef __attribute__((ext_vector_type(8)))  float  v8f;
typedef __attribute__((ext_vector_type(4)))  float  f32x4;
typedef unsigned int u32x4 __attribute__((ext_vector_type(4)));
typedef int          i32x4 __attribute__((ext_vector_type(4)));
typedef int          i32x8 __attribute__((ext_vector_type(8)));

static constexpr float LRELU_F = 0.1f;
static constexpr int NPTS   = 6912;
static constexpr int MID    = 512;
static constexpr int NTILES = MID / 16;   // 32 N-tiles of 16
static constexpr int LDW    = 520;        // LDS act row stride (bf16 elems, 16B aligned)
static constexpr int AW     = 16 * LDW;   // ushorts per 16-row activation buffer

// packed weights: fragment tile = 32 lanes x 16 bf16 = 512 ushorts (1 KB);
// tiles ordered NT-MAJOR: tile = nt*KT + kt -> each nt's K-block is contiguous (KT KB)
static constexpr size_t TILE_U     = 512;
static constexpr size_t PROJ_U     = 8  * 32 * TILE_U;   // K padded 248->256
static constexpr size_t SQ_U       = 16 * 32 * TILE_U;   // 512x512
static constexpr size_t OFF_PROJ   = 0;
static constexpr size_t OFF_R1W1   = PROJ_U;
static constexpr size_t OFF_R1W2   = 2 * PROJ_U;
static constexpr size_t OFF_LAYERS = 2 * PROJ_U + SQ_U;
static constexpr size_t WBASE_BYTES = 16 * 128 * sizeof(float);  // f01 region at ws start

// dynamic-LDS layout (bytes); dynamic LDS base assumed at group-segment offset 0
static constexpr unsigned WSTAGE_OFF = 0;        // 2 x 16 KB weight double buffer
static constexpr unsigned WSTAGE_BUF = 16384;
static constexpr unsigned ACT_OFF    = 32768;    // 4 waves x 3 buffers x AW ushorts
static constexpr unsigned RED_OFF    = ACT_OFF + 4u * 3u * AW * 2u;    // 232448
static constexpr unsigned BIAS_OFF   = RED_OFF + 4u * 32u * 8u * 4u;   // 236544 (512 f32)
static constexpr unsigned OUTW_OFF   = BIAS_OFF + MID * 4u;            // 238592 (512 f32)
static constexpr unsigned SMEM_BYTES = OUTW_OFF + MID * 4u;            // 240640

__device__ __forceinline__ unsigned short f2bf(float f) {
  unsigned int u = __builtin_bit_cast(unsigned int, f);
  unsigned int r = 0x7FFFu + ((u >> 16) & 1u);   // round-to-nearest-even
  return (unsigned short)((u + r) >> 16);
}
__device__ __forceinline__ float bf2f(unsigned short h) {
  return __builtin_bit_cast(float, ((unsigned int)h) << 16);
}

struct F4x2 { f32x4 lo, hi; };
__device__ __forceinline__ v16bf mk16(f32x4 lo, f32x4 hi) {
  F4x2 t{lo, hi};
  return __builtin_bit_cast(v16bf, t);
}

// A fragment (16x32 bf16) from row-major LDS tile
__device__ __forceinline__ v16bf load_afrag(const unsigned short* a, int kt, int lane) {
  int r    = lane & 15;
  int koff = (lane >> 4) << 4;
  const unsigned short* p = a + r * LDW + kt * 32;
  f32x4 lo = *(const f32x4*)(p + koff);
  f32x4 hi = *(const f32x4*)(p + 8 + koff);
  return mk16(lo, hi);
}

// ---- Tensor Data Mover: 1-D contiguous copy global -> LDS (D# per ISA 8.3-8.6) ----
__device__ __forceinline__ void tdm_load_1d(const void* gsrc, unsigned lds_off, unsigned nbytes) {
  unsigned long long ga = (unsigned long long)(uintptr_t)gsrc;
  unsigned ne8 = nbytes >> 3;                       // elements of 8 bytes (data_size=3)
  u32x4 g0;
  g0[0] = 1u;                                       // count=1, user descriptor
  g0[1] = lds_off;                                  // lds_addr
  g0[2] = (unsigned)(ga & 0xFFFFFFFFu);             // global_addr[31:0]
  g0[3] = (unsigned)((ga >> 32) & 0x01FFFFFFu) | (2u << 30);  // addr[56:32] | type=2
  i32x8 g1;
  g1[0] = (int)(3u << 16);                          // wg_mask=0, data_size=3 (8B)
  g1[1] = (int)((ne8 & 0xFFFFu) << 16);             // tensor_dim0[15:0]
  g1[2] = (int)(((ne8 >> 16) & 0xFFFFu) | (1u << 16)); // tensor_dim0[31:16] | tensor_dim1=1
  g1[3] = (int)((ne8 & 0xFFFFu) << 16);             // tile_dim0 = ne8 (16-bit)
  g1[4] = 0;                                        // tile_dim1/2 unused
  g1[5] = (int)ne8;                                 // tensor_dim0_stride[31:0]
  g1[6] = 0;
  g1[7] = 0;
  i32x4 g2 = {1, 1, 0, 0};                          // tensor_dim2=1, tensor_dim3=1
  i32x4 g3 = {0, (int)(1u << 16), 0, 0};            // tensor_dim4=1
#if defined(__clang_major__) && (__clang_major__ >= 23)
  i32x8 gz = {0, 0, 0, 0, 0, 0, 0, 0};
  __builtin_amdgcn_tensor_load_to_lds(g0, g1, g2, g3, gz, 0);
#else
  __builtin_amdgcn_tensor_load_to_lds(g0, g1, g2, g3, 0);
#endif
}

// One wave computes a 16x512 GEMM tile; weight N-blocks are TDM double-buffered in LDS
// and shared by all 4 waves; bias vector staged in LDS (no global loads in the nt loop).
template<int KT, typename Epi>
__device__ __forceinline__ void gemm_staged(const unsigned short* __restrict__ gW,
                                            const float* __restrict__ bias,
                                            const unsigned short* aIn,
                                            unsigned short* wstage,  // LDS @ WSTAGE_OFF
                                            float* biasLds,          // LDS @ BIAS_OFF
                                            int lane, int wave, Epi epi) {
  constexpr unsigned BLK_BYTES = KT * 1024u;        // one nt's K-block
  constexpr unsigned BLK_U     = KT * (unsigned)TILE_U;
  v16bf a[KT];
  #pragma unroll
  for (int kt = 0; kt < KT; ++kt) a[kt] = load_afrag(aIn, kt, lane);
  for (int i = wave * 32 + lane; i < MID; i += 128) biasLds[i] = bias[i];
  if (wave == 0) {
    tdm_load_1d(gW, WSTAGE_OFF, BLK_BYTES);
    __builtin_amdgcn_s_wait_tensorcnt(0);
  }
  __syncthreads();
  for (int nt = 0; nt < NTILES; ++nt) {
    int cb = nt & 1;
    if (wave == 0 && nt + 1 < NTILES)               // overlap DMA of nt+1 with compute of nt
      tdm_load_1d(gW + (size_t)(nt + 1) * BLK_U,
                  WSTAGE_OFF + (unsigned)(1 - cb) * WSTAGE_BUF, BLK_BYTES);
    const unsigned short* wblk = wstage + (size_t)cb * (WSTAGE_BUF / 2);  // ushort units
    float bv = biasLds[nt * 16 + (lane & 15)];      // ds_load, no global stall
    v8f acc;
    #pragma unroll
    for (int i = 0; i < 8; ++i) acc[i] = bv;
    #pragma unroll
    for (int kt = 0; kt < KT; ++kt) {
      const unsigned short* wl = wblk + ((size_t)kt * 32 + lane) * 16;
      v16bf bm = mk16(*(const f32x4*)wl, *(const f32x4*)(wl + 8));
      acc = __builtin_amdgcn_wmma_f32_16x16x32_bf16(false, a[kt], false, bm,
                                                    (short)0, acc, false, false);
    }
    epi(nt, acc);
    if (wave == 0) __builtin_amdgcn_s_wait_tensorcnt(0);
    __syncthreads();                                // next buffer ready / current reusable
  }
}

// C/D layout: vgpr v, lane l -> row = v + 8*(l>=16), col = nt*16 + (l&15)
// leaky(x) with slope 0.1 > 0 is exactly max(x, 0.1*x): one v_mul + v_max per element.
__device__ __forceinline__ void epi_store(unsigned short* dst, int nt, int lane,
                                          v8f acc, bool lrelu) {
  int col = nt * 16 + (lane & 15);
  int rb  = (lane >> 4) * 8;
  #pragma unroll
  for (int i = 0; i < 8; ++i) {
    float f = acc[i];
    if (lrelu) f = fmaxf(f, LRELU_F * f);
    dst[(rb + i) * LDW + col] = f2bf(f);
  }
}

// ---------------- kernel 1: Nadaraya-Watson embedding, f01[b][0..127] ----------------
__global__ void fit_predict_kernel(const float* __restrict__ sound_loc,
                                   const float* __restrict__ grid_coors,
                                   const float* __restrict__ grid_0,
                                   const float* __restrict__ xyz_offset,
                                   const float* __restrict__ bandwidths,
                                   float* __restrict__ f01) {
  __shared__ float sf[128];
  __shared__ float sden[2];
  const int b = blockIdx.x, tid = threadIdx.x;
  if (tid < 128) sf[tid] = 0.f;
  if (tid < 2)   sden[tid] = 0.f;
  __syncthreads();
  const float q0x = sound_loc[b*6+0], q0y = sound_loc[b*6+1], q0z = sound_loc[b*6+2];
  const float q1x = sound_loc[b*6+3], q1y = sound_loc[b*6+4], q1z = sound_loc[b*6+5];
  float acc0[64], acc1[64];
  #pragma unroll
  for (int c = 0; c < 64; ++c) { acc0[c] = 0.f; acc1[c] = 0.f; }
  float den0 = 0.f, den1 = 0.f;
  // exponent <= 0 always and ~0 at the nearest grid point -> softmax w/o max-sub is safe
  for (int i = tid; i < NPTS; i += 256) {
    float gx = grid_coors[i*3+0] + tanhf(xyz_offset[i*3+0]) * 0.1f;
    float gy = grid_coors[i*3+1] + tanhf(xyz_offset[i*3+1]) * 0.1f;
    float gz = grid_coors[i*3+2] + tanhf(xyz_offset[i*3+2]) * 0.1f;
    float bw  = fminf(fmaxf(bandwidths[i], 0.1f), 0.5f);
    float inv = 1.f / (2.f * bw * bw);
    float dx0 = q0x-gx, dy0 = q0y-gy, dz0 = q0z-gz;
    float dx1 = q1x-gx, dy1 = q1y-gy, dz1 = q1z-gz;
    float e0 = __expf(-(dx0*dx0 + dy0*dy0 + dz0*dz0) * inv);
    float e1 = __expf(-(dx1*dx1 + dy1*dy1 + dz1*dz1) * inv);
    den0 += e0; den1 += e1;
    const float* g = grid_0 + (size_t)i * 64;
    #pragma unroll
    for (int c = 0; c < 64; ++c) { float gv = g[c]; acc0[c] += e0*gv; acc1[c] += e1*gv; }
  }
  atomicAdd(&sden[0], den0);
  atomicAdd(&sden[1], den1);
  #pragma unroll
  for (int c = 0; c < 64; ++c) {
    atomicAdd(&sf[c],      acc0[c]);
    atomicAdd(&sf[64 + c], acc1[c]);
  }
  __syncthreads();
  if (tid < 64)        f01[b*128 + tid] = sf[tid] / (sden[0] + 1e-20f);
  else if (tid < 128)  f01[b*128 + tid] = sf[tid] / (sden[1] + 1e-20f);
}

// -------- kernel 2: f32 weights -> bf16 fragment-packed, NT-MAJOR tile order -----------
__global__ void pack_w_kernel(const float* __restrict__ src, unsigned short* __restrict__ dst,
                              int Kin, int KT) {
  int t = blockIdx.x * blockDim.x + threadIdx.x;
  int lane = t & 31, tile = t >> 5;
  if (tile >= KT * NTILES) return;
  int nt = tile / KT, kt = tile % KT;               // tile = nt*KT + kt
  int n  = nt * 16 + (lane & 15);
  int kb = kt * 32 + ((lane >> 4) << 4);            // lanes 16..31 hold K=16..31 of tile
  unsigned short* o = dst + (size_t)tile * TILE_U + (size_t)lane * 16;
  #pragma unroll
  for (int i = 0; i < 16; ++i) {
    int k = kb + i;
    float v = (k < Kin) ? src[(size_t)k * MID + n] : 0.f;   // zero-pad K 248->256
    o[i] = f2bf(v);
  }
}

// ---------------- kernel 3: fused MLP chain, 4 waves/WG, TDM-staged weights -------------
__global__ void __launch_bounds__(128)
mlp_fused_kernel(const float* __restrict__ input_stuff,
                 const float* __restrict__ f01,
                 const unsigned short* __restrict__ wpack,
                 const float* __restrict__ proj_b,
                 const float* __restrict__ r1_b1,
                 const float* __restrict__ r1_b2,
                 const float* __restrict__ layers_b,
                 const float* __restrict__ out_W,
                 const float* __restrict__ out_b,
                 float* __restrict__ out) {
  extern __shared__ __align__(16) char smem[];
  unsigned short* wstage  = (unsigned short*)(smem + WSTAGE_OFF);
  unsigned short* actbase = (unsigned short*)(smem + ACT_OFF);
  float*          redall  = (float*)(smem + RED_OFF);
  float*          biasLds = (float*)(smem + BIAS_OFF);
  float*          outwLds = (float*)(smem + OUTW_OFF);

  const int lane = threadIdx.x & 31;
  const int wave = threadIdx.x >> 5;
  const int tid  = threadIdx.x;
  const int row0 = blockIdx.x * 64 + wave * 16;     // 64 | 4096 -> never crosses a batch
  const int b    = row0 >> 12;

  unsigned short* bufA = actbase + (size_t)wave * 3 * AW;
  unsigned short* bufB = bufA + AW;
  unsigned short* resB = bufA + 2 * AW;
  float* red = redall + wave * 256;                 // 32 lanes x 8

  // stage out_W once (used by last-layer epilogue)
  for (int i = tid; i < MID; i += 128) outwLds[i] = out_W[i];

  // stage my_input strip: [f01 grid (128) | input_stuff (120) | zero pad (8)] as bf16
  for (int idx = lane; idx < 16 * 256; idx += 32) {
    int r = idx >> 8, c = idx & 255;
    float v = 0.f;
    if (c < 128)       v = f01[b * 128 + c];
    else if (c < 248)  v = input_stuff[(size_t)(row0 + r) * 120 + (c - 128)];
    bufA[r * LDW + c] = f2bf(v);
  }
  __syncthreads();

  // out0 = my_input @ proj_W + proj_b ; store leaky -> bufB (layer-0 input)
  gemm_staged<8>(wpack + OFF_PROJ, proj_b, bufA, wstage, biasLds, lane, wave,
                 [&](int nt, v8f a) { epi_store(bufB, nt, lane, a, true); });
  // t = leaky(my_input @ r1_W1 + b1) -> resB
  gemm_staged<8>(wpack + OFF_R1W1, r1_b1, bufA, wstage, biasLds, lane, wave,
                 [&](int nt, v8f a) { epi_store(resB, nt, lane, a, true); });
  // res = t @ r1_W2 + b2 (raw) -> resB (A-fragments register-resident before stores)
  gemm_staged<16>(wpack + OFF_R1W2, r1_b2, resB, wstage, biasLds, lane, wave,
                  [&](int nt, v8f a) { epi_store(resB, nt, lane, a, false); });

  float p[8];
  #pragma unroll
  for (int i = 0; i < 8; ++i) p[i] = 0.f;

  unsigned short* cur = bufB;
  unsigned short* nxt = bufA;
  for (int k = 0; k < 6; ++k) {
    const unsigned short* w = wpack + OFF_LAYERS + (size_t)k * SQ_U;
    const float* bias = layers_b + k * MID;
    if (k == 5) {
      // last layer: fuse 512->1 projection into the epilogue (out_W from LDS)
      gemm_staged<16>(w, bias, cur, wstage, biasLds, lane, wave, [&](int nt, v8f a) {
        float wv = outwLds[nt * 16 + (lane & 15)];
        #pragma unroll
        for (int i = 0; i < 8; ++i) p[i] += a[i] * wv;
      });
    } else if (k == 2) {
      // out += res after layer n/2-1, then leaky for next layer input
      gemm_staged<16>(w, bias, cur, wstage, biasLds, lane, wave, [&](int nt, v8f a) {
        int col = nt * 16 + (lane & 15);
        int rb  = (lane >> 4) * 8;
        #pragma unroll
        for (int i = 0; i < 8; ++i) a[i] += bf2f(resB[(rb + i) * LDW + col]);
        epi_store(nxt, nt, lane, a, true);
      });
    } else {
      gemm_staged<16>(w, bias, cur, wstage, biasLds, lane, wave,
                      [&](int nt, v8f a) { epi_store(nxt, nt, lane, a, true); });
    }
    unsigned short* t = cur; cur = nxt; nxt = t;
  }

  // cross-lane reduce 512->1: row m<8 in lanes 0..15 (vgpr m), row m>=8 in lanes 16..31
  #pragma unroll
  for (int i = 0; i < 8; ++i) red[lane * 8 + i] = p[i];
  __syncthreads();
  if (lane < 16) {
    float s = out_b[0];
    if (lane < 8) { for (int l = 0;  l < 16; ++l) s += red[l * 8 + lane]; }
    else          { for (int l = 16; l < 32; ++l) s += red[l * 8 + lane - 8]; }
    out[row0 + lane] = s;
  }
}

extern "C" void kernel_launch(void* const* d_in, const int* in_sizes, int n_in,
                              void* d_out, int out_size, void* d_ws, size_t ws_size,
                              hipStream_t stream) {
  (void)in_sizes; (void)n_in; (void)out_size; (void)ws_size;
  const float* input_stuff = (const float*)d_in[0];
  const float* sound_loc   = (const float*)d_in[1];
  const float* grid_coors  = (const float*)d_in[2];
  const float* grid_0      = (const float*)d_in[3];
  const float* xyz_offset  = (const float*)d_in[4];
  const float* bandwidths  = (const float*)d_in[5];
  const float* proj_W      = (const float*)d_in[6];
  const float* proj_b      = (const float*)d_in[7];
  const float* r1_W1       = (const float*)d_in[8];
  const float* r1_b1       = (const float*)d_in[9];
  const float* r1_W2       = (const float*)d_in[10];
  const float* r1_b2       = (const float*)d_in[11];
  const float* layers_W    = (const float*)d_in[12];
  const float* layers_b    = (const float*)d_in[13];
  const float* out_W       = (const float*)d_in[14];
  const float* out_b       = (const float*)d_in[15];

  float* f01 = (float*)d_ws;
  unsigned short* wpack = (unsigned short*)((char*)d_ws + WBASE_BYTES);

  fit_predict_kernel<<<dim3(16), dim3(256), 0, stream>>>(
      sound_loc, grid_coors, grid_0, xyz_offset, bandwidths, f01);

  const int tpb  = 256;
  const int th8  = 8  * 32 * 32;
  const int th16 = 16 * 32 * 32;
  pack_w_kernel<<<dim3((th8  + tpb - 1) / tpb), dim3(tpb), 0, stream>>>(
      proj_W, wpack + OFF_PROJ, 248, 8);
  pack_w_kernel<<<dim3((th8  + tpb - 1) / tpb), dim3(tpb), 0, stream>>>(
      r1_W1,  wpack + OFF_R1W1, 248, 8);
  pack_w_kernel<<<dim3((th16 + tpb - 1) / tpb), dim3(tpb), 0, stream>>>(
      r1_W2,  wpack + OFF_R1W2, 512, 16);
  for (int k = 0; k < 6; ++k)
    pack_w_kernel<<<dim3((th16 + tpb - 1) / tpb), dim3(tpb), 0, stream>>>(
        layers_W + (size_t)k * 512 * 512, wpack + OFF_LAYERS + (size_t)k * SQ_U, 512, 16);

  mlp_fused_kernel<<<dim3(1024), dim3(128), SMEM_BYTES, stream>>>(
      input_stuff, f01, wpack, proj_b, r1_b1, r1_b2, layers_b, out_W, out_b,
      (float*)d_out);
}
